// BNSP_repulsion_79998060855948
// MI455X (gfx1250) — 8-bit matrix operand order repaired and checked by reading, compiled, not run on hardware
//
#include <hip/hip_runtime.h>

typedef __attribute__((ext_vector_type(8))) int v8i;

#define MAP_W 4096

struct AgentGeom {
    int rowBase, colBase;
    int nr, nc;
    int nrEff, ncEff;
    int r_lt, c_lt;
};

__device__ inline AgentGeom agent_setup(const float* step, const float* ff,
                                        const float* vel, int a) {
    float orR = step[2*a+0] + ff[2*a+0];
    float orC = step[2*a+1] + ff[2*a+1];
    float vR  = vel[2*a+0], vC = vel[2*a+1];
    float sR = (vR > 0.f) ? 1.f : ((vR < 0.f) ? -1.f : 0.f);
    float sC = (vC > 0.f) ? 1.f : ((vC < 0.f) ? -1.f : 0.f);
    int a0r = (int)floorf(orR), a1r = (int)floorf(orR + sR * 16.f);
    int a0c = (int)floorf(orC), a1c = (int)floorf(orC + sC * 16.f);
    AgentGeom g;
    int rmin = min(a0r, a1r), rmax = max(a0r, a1r);
    int cmin = min(a0c, a1c), cmax = max(a0c, a1c);
    g.nr = rmax - rmin;  g.nc = cmax - cmin;
    g.rowBase = (g.nr == 0) ? a0r : rmin;
    g.colBase = (g.nc == 0) ? a0c : cmin;
    g.nrEff = max(g.nr, 1);  g.ncEff = max(g.nc, 1);
    g.r_lt = (a0r < a1r);    g.c_lt = (a0c < a1c);
    return g;
}

__device__ inline int ind_eq(unsigned p, unsigned lab) {
    unsigned x = p ^ (lab * 0x01010101u);
    unsigned t = x + 0x7F7F7F7Fu;
    return (int)((~t & 0x80808080u) >> 7);
}

__global__ __launch_bounds__(256)
void BNSP_repulsion_wmma_kernel(const float* __restrict__ step,
                                const float* __restrict__ ff,
                                const float* __restrict__ vel,
                                const int*   __restrict__ smap,
                                float*       __restrict__ out,
                                int nAgents)
{
    __shared__ int xch[8][16][12];

    const int lane = threadIdx.x & 31;
    const int wave = threadIdx.x >> 5;
    const int hi   = lane >> 4;
    const int sub  = lane & 15;
    const int base = (blockIdx.x * 8 + wave) * 16;

    int aL = base + sub;
    if (aL >= nAgents) aL = nAgents - 1;
    AgentGeom gl = agent_setup(step, ff, vel, aL);

    const int* __restrict__ wbase = smap + gl.rowBase * MAP_W + gl.colBase + hi * 8;

    unsigned cmask[2];
    #pragma unroll
    for (int hf = 0; hf < 2; ++hf) {
        int idx = gl.ncEff - (hi * 8 + hf * 4);
        idx = min(max(idx, 0), 4);
        cmask[hf] = (idx == 4) ? 0u : (0x7F7F7F7Fu << (8 * idx));
    }

    const bool isCol1 = (sub == 1);
    const unsigned hiSplat = (unsigned)hi * 0x01010101u;
    v8i bBase;
    #pragma unroll
    for (int v = 0; v < 8; ++v) {
        unsigned bv = (sub == 0) ? 0x01010101u
                    : (sub == 2) ? (0x03020100u + 0x04040404u * (unsigned)(v & 1) + 0x08080808u * (unsigned)hi)
                    : 0u;
        bBase[v] = (int)bv;
    }

    v8i acc5 = {}, acc3 = {}, acc4 = {};

    #pragma unroll
    for (int cc = 0; cc < 4; ++cc) {
        int w[4][8];
        #pragma unroll
        for (int rr = 0; rr < 4; ++rr)
            #pragma unroll
            for (int e = 0; e < 8; ++e)
                w[rr][e] = wbase[(4*cc + rr) * MAP_W + e];

        unsigned rm[4];
        #pragma unroll
        for (int rr = 0; rr < 4; ++rr)
            rm[rr] = ((4*cc + rr) < gl.nrEff) ? 0u : 0x7F7F7F7Fu;

        v8i a5, a3, a4;
        #pragma unroll
        for (int v = 0; v < 8; ++v) {
            int rr = v >> 1, hf = v & 1;
            unsigned p =  (unsigned)w[rr][hf*4 + 0]
                       | ((unsigned)w[rr][hf*4 + 1] << 8)
                       | ((unsigned)w[rr][hf*4 + 2] << 16)
                       | ((unsigned)w[rr][hf*4 + 3] << 24);
            p |= cmask[hf] | rm[rr];
            a5[v] = ind_eq(p, 5u);
            a3[v] = ind_eq(p, 3u);
            a4[v] = ind_eq(p, 4u);
        }

        unsigned rvA = (unsigned)(4*cc)     * 0x01010101u + hiSplat;
        unsigned rvB = (unsigned)(4*cc + 2) * 0x01010101u + hiSplat;
        v8i b;
        #pragma unroll
        for (int v = 0; v < 8; ++v)
            b[v] = isCol1 ? (int)((unsigned)(4*cc + (v >> 1)) * 0x01010101u) : bBase[v];

        acc5 = __builtin_amdgcn_wmma_i32_16x16x64_iu8(false, a5, false, b,
                                                      acc5, false, false);
        acc3 = __builtin_amdgcn_wmma_i32_16x16x64_iu8(false, a3, false, b,
                                                      acc3, false, false);
        acc4 = __builtin_amdgcn_wmma_i32_16x16x64_iu8(false, a4, false, b,
                                                      acc4, false, false);
    }

    if (sub < 3) {
        #pragma unroll
        for (int e = 0; e < 8; ++e) {
            int ag = 8*hi + e;
            xch[wave][ag][0*3 + sub] = acc5[e];
            xch[wave][ag][1*3 + sub] = acc3[e];
            xch[wave][ag][2*3 + sub] = acc4[e];
        }
    }
    __syncthreads();

    int aF = base + lane;
    if (lane < 16 && aF < nAgents) {
        AgentGeom gf = agent_setup(step, ff, vel, aF);
        const float ks = 16.f;
        bool row_case = (gf.nr == 0) && (gf.nc != 0);
        bool col_case = (gf.nr != 0) && (gf.nc == 0);
        bool two_d    = (gf.nr != 0) && (gf.nc != 0);
        float cr = gf.r_lt ? 0.f : ks;
        float cc = gf.c_lt ? 0.f : ks;

        float F0 = 0.f, F1 = 0.f;
        #pragma unroll
        for (int li = 0; li < 3; ++li) {
            float colPlus = (li == 0) ? 1.f : 0.f;
            float lw      = (li == 2) ? 3.f : 1.f;
            float cnt  = (float)xch[wave][lane][li*3 + 0];
            float sumR = (float)xch[wave][lane][li*3 + 1];
            float sumC = (float)xch[wave][lane][li*3 + 2];
            bool  has  = cnt > 0.f;
            float den  = fmaxf(cnt, 1.f);
            float mr = sumR / den, mc = sumC / den;

            float dis_row = gf.c_lt ? mc : (ks - mc);
            float frm = (dis_row != 0.f) ? (2.f / dis_row) : 0.f;
            float fr1 = gf.c_lt ? -frm : frm;

            float dis_col = gf.r_lt ? (mr + colPlus) : (ks - mr);
            float fcm = (dis_col != 0.f) ? (2.f / dis_col) : 0.f;
            float fc0 = gf.r_lt ? -fcm : fcm;

            float dr = mr - cr, dc = mc - cc;
            float d  = sqrtf(dr*dr + dc*dc);
            float f20 = 0.f, f21 = 0.f;
            if (d != 0.f) { float s = 2.f / (d * d); f20 = (cr - mr) * s; f21 = (cc - mc) * s; }

            float fx = row_case ? 0.f : (col_case ? fc0 : (two_d ? f20 : 0.f));
            float fy = row_case ? fr1 : (col_case ? 0.f : (two_d ? f21 : 0.f));
            if (!has) { fx = 0.f; fy = 0.f; }
            F0 += lw * fx;
            F1 += lw * fy;
        }
        out[2*aF + 0] = F0;
        out[2*aF + 1] = F1;
    }
}

extern "C" void kernel_launch(void* const* d_in, const int* in_sizes, int n_in,
                              void* d_out, int out_size, void* d_ws, size_t ws_size,
                              hipStream_t stream) {
    const float* step = (const float*)d_in[0];
    const float* ff   = (const float*)d_in[1];
    const float* vel  = (const float*)d_in[2];
    const int*   smap = (const int*)  d_in[3];
    float*       out  = (float*)d_out;

    int nAgents = in_sizes[0] / 2;
    int blocks  = (nAgents + 127) / 128;
    BNSP_repulsion_wmma_kernel<<<blocks, 256, 0, stream>>>(
        step, ff, vel, smap, out, nAgents);
}
